// DABM_28544352649780
// MI455X (gfx1250) — matrix operand order repaired and checked by reading, not compiled, not run on hardware
//
#include <hip/hip_runtime.h>
#include <hip/hip_bf16.h>

// ---- compile-time probes (stderr): which CDNA5 builtins exist here ----
#if __has_builtin(__builtin_amdgcn_global_load_async_to_lds_b32)
#warning PROBE: has __builtin_amdgcn_global_load_async_to_lds_b32
#endif
#if __has_builtin(__builtin_amdgcn_global_load_async_to_lds_b128)
#warning PROBE: has __builtin_amdgcn_global_load_async_to_lds_b128
#endif
#if __has_builtin(__builtin_amdgcn_s_wait_asynccnt)
#warning PROBE: has __builtin_amdgcn_s_wait_asynccnt
#endif
#if __has_builtin(__builtin_amdgcn_tensor_load_to_lds)
#warning PROBE: has __builtin_amdgcn_tensor_load_to_lds
#endif

typedef __attribute__((ext_vector_type(16))) _Float16 v16h;
typedef __attribute__((ext_vector_type(8)))  float    v8f;

static constexpr int Bn = 8, Cn = 128, Hn = 256, Wn = 256;
static constexpr int HW = Hn * Wn;          // 65536
static constexpr int CR = 8;                // C / RED

// ---- CDNA5 async copy global->LDS (ASYNCcnt path), inline asm ----
__device__ __forceinline__ void async_copy_b32(unsigned lds_byte_off, const float* g) {
  asm volatile("global_load_async_to_lds_b32 %0, %1, off"
               :: "v"(lds_byte_off),
                  "v"((unsigned long long)(uintptr_t)g)
               : "memory");
}
__device__ __forceinline__ void wait_async0() {
  asm volatile("s_wait_asynccnt 0" ::: "memory");
}

// ---------------------------------------------------------------------------
// Kernel 1: global average pool  x[B,C,H,W] -> pooled[B,C]
// ---------------------------------------------------------------------------
__global__ void gap_kernel(const float* __restrict__ x, float* __restrict__ pooled) {
  const int bc = blockIdx.x;                       // 0 .. B*C-1
  const float4* p4 = (const float4*)(x + (size_t)bc * HW);
  float s = 0.f;
  for (int i = threadIdx.x; i < HW / 4; i += 256) {
    float4 v = p4[i];
    s += (v.x + v.y) + (v.z + v.w);
  }
  __shared__ float red[256];
  red[threadIdx.x] = s;
  __syncthreads();
  for (int off = 128; off > 0; off >>= 1) {
    if (threadIdx.x < (unsigned)off) red[threadIdx.x] += red[threadIdx.x + off];
    __syncthreads();
  }
  if (threadIdx.x == 0) pooled[bc] = red[0] * (1.0f / HW);
}

// ---------------------------------------------------------------------------
// Kernel 2: channel-attention MLP via WMMA (one wave32)
//   h  = relu(pooled @ W1^T + b1)    [8,8]   (GEMM1: M=8pad16, N=8pad16, K=128)
//   ca = sigmoid(h @ W2^T + b2)      [8,128] (GEMM2: M=8pad16, N=128, K=8pad32)
// ---------------------------------------------------------------------------
__global__ void ca_mlp_kernel(const float* __restrict__ pooled,
                              const float* __restrict__ w1p, const float* __restrict__ b1p,
                              const float* __restrict__ w2p, const float* __restrict__ b2p,
                              float* __restrict__ ca) {
  const int L   = threadIdx.x;     // 0..31, full wave (EXEC all ones for WMMA)
  const int n   = L & 15;
  const int hi8 = (L >> 4) * 8;    // 0 or 8
  const int hi16 = (L >> 4) * 16;  // 0 or 16
  __shared__ float hmat[16 * 16];

  // ---- GEMM1: D[m][n] = sum_k pooled[m][k] * W1[n][k] ----
  v8f acc = {};
  for (int k0 = 0; k0 < 128; k0 += 32) {
    v16h a, b;
#pragma unroll
    for (int t = 0; t < 16; ++t) {
      const int v = t >> 1, p = t & 1;
      const int ka = (v < 4 ? 0 : 16) + hi8 + (v & 3) * 2 + p;   // K index for A
      a[t] = (_Float16)((n < 8) ? pooled[n * 128 + k0 + ka] : 0.f);
      const int kb = (t < 8 ? 0 : 8) + hi8 + t;                   // K index for B
      b[t] = (_Float16)((n < 8) ? w1p[n * 128 + k0 + kb] : 0.f);
    }
    acc = __builtin_amdgcn_wmma_f32_16x16x32_f16(false, a, false, b,
                                                 (short)0, acc, false, false);
  }
#pragma unroll
  for (int r = 0; r < 8; ++r) {
    const int m = r + hi8;
    float v = acc[r] + ((n < 8) ? b1p[n] : 0.f);
    hmat[m * 16 + n] = v > 0.f ? v : 0.f;
  }
  __syncthreads();

  // ---- GEMM2: D[m][c] = sum_j h[m][j] * W2[c][j], K=8 padded to 32 ----
  v16h a2;
#pragma unroll
  for (int t = 0; t < 16; ++t) {
    const int v = t >> 1, p = t & 1;
    const int ka = (v < 4 ? 0 : 16) + hi8 + (v & 3) * 2 + p;
    a2[t] = (ka < 8) ? (_Float16)hmat[n * 16 + ka] : (_Float16)0.f;
  }
#pragma unroll
  for (int tile = 0; tile < 8; ++tile) {
    const int c = tile * 16 + n;
    v16h b2;
#pragma unroll
    for (int t = 0; t < 16; ++t) {
      const int kb = (t < 8 ? 0 : 8) + hi8 + t;
      b2[t] = (kb < 8) ? (_Float16)w2p[c * CR + kb] : (_Float16)0.f;
    }
    v8f acc2 = {};
    acc2 = __builtin_amdgcn_wmma_f32_16x16x32_f16(false, a2, false, b2,
                                                  (short)0, acc2, false, false);
#pragma unroll
    for (int r = 0; r < 8; ++r) {
      const int m = r + hi8;
      if (m < 8) {
        const float z = acc2[r] + b2p[c];
        ca[m * Cn + c] = 1.0f / (1.0f + __expf(-z));
      }
    }
  }
}

// ---------------------------------------------------------------------------
// Kernel 3: 3x3 conv (C=128 -> 2) -> unnormalized sample coords (ix,iy)
// 16x16 pixel tile per block, 18x18 halo, 16-channel chunks.
// Double-buffered LDS staging via GLOBAL_LOAD_ASYNC_TO_LDS_B32:
//   iter k: wait(asyncs of chunk k) ; barrier ; issue asyncs chunk k+1 into
//   other buffer ; compute chunk k (overlapped with in-flight copies).
// ---------------------------------------------------------------------------
static constexpr int TILE_ELEMS = 16 * 18 * 18;   // 5184 floats per chunk

__device__ __forceinline__ void stage_chunk(const float* __restrict__ x,
                                            float* __restrict__ buf,   // LDS
                                            int b, int c0, int bx, int by, int tid) {
  for (int i = tid; i < TILE_ELEMS; i += 256) {
    const int c  = i / (18 * 18);
    const int rr = (i % (18 * 18)) / 18;
    const int qq = i % 18;
    const int gy = by + rr - 1, gx = bx + qq - 1;
    if (gy >= 0 && gy < Hn && gx >= 0 && gx < Wn) {
      const float* g = &x[(((size_t)b * Cn + c0 + c) * Hn + gy) * Wn + gx];
      async_copy_b32((unsigned)(uintptr_t)&buf[i], g);   // cache -> LDS, no VGPR
    } else {
      buf[i] = 0.f;                                      // halo zero-fill
    }
  }
}

__global__ void offset_kernel(const float* __restrict__ x,
                              const float* __restrict__ wsp,
                              const float* __restrict__ bsp,
                              float* __restrict__ coords) {
  __shared__ float xt[2][TILE_ELEMS];   // double buffer: 2 x 16ch x 18x18
  __shared__ float wl[2 * Cn * 9];      // all conv weights [o][c][3][3]

  const int tx = threadIdx.x, ty = threadIdx.y;
  const int tid = ty * 16 + tx;
  const int bx = blockIdx.x * 16, by = blockIdx.y * 16;
  const int b = blockIdx.z;

  for (int i = tid; i < 2 * Cn * 9; i += 256) wl[i] = wsp[i];

  stage_chunk(x, xt[0], b, 0, bx, by, tid);   // prologue: chunk 0 -> buf 0

  float a0 = 0.f, a1 = 0.f;
  for (int chunk = 0; chunk < 8; ++chunk) {
    wait_async0();          // this wave's copies for current buffer are in LDS
    __syncthreads();        // all waves' copies done; previous compute done

    if (chunk < 7)
      stage_chunk(x, xt[(chunk + 1) & 1], b, (chunk + 1) * 16, bx, by, tid);

    const float* cur = xt[chunk & 1];
    const int c0 = chunk * 16;
#pragma unroll 4
    for (int c = 0; c < 16; ++c) {
      const float* wr0 = &wl[(c0 + c) * 9];
      const float* wr1 = &wl[Cn * 9 + (c0 + c) * 9];
      const float* xc = &cur[c * (18 * 18)];
#pragma unroll
      for (int ky = 0; ky < 3; ++ky) {
#pragma unroll
        for (int kx = 0; kx < 3; ++kx) {
          const float v = xc[(ty + ky) * 18 + (tx + kx)];
          a0 = fmaf(wr0[ky * 3 + kx], v, a0);
          a1 = fmaf(wr1[ky * 3 + kx], v, a1);
        }
      }
    }
  }
  const int w = bx + tx, h = by + ty;
  const float gx = (-1.0f + (2.0f / (Wn - 1)) * w) + a0 + bsp[0];
  const float gy = (-1.0f + (2.0f / (Hn - 1)) * h) + a1 + bsp[1];
  const float ix = ((gx + 1.0f) * Wn - 1.0f) * 0.5f;   // align_corners=False
  const float iy = ((gy + 1.0f) * Hn - 1.0f) * 0.5f;
  const size_t p = ((size_t)b * HW + (size_t)h * Wn + w) * 2;
  coords[p + 0] = ix;
  coords[p + 1] = iy;
}

// ---------------------------------------------------------------------------
// Kernel 4: fused epilogue  out = x*ca + grid_sample_bilinear(x, coords)
// ---------------------------------------------------------------------------
__global__ void fuse_kernel(const float* __restrict__ x,
                            const float* __restrict__ ca,
                            const float* __restrict__ coords,
                            float* __restrict__ out) {
  const int b = blockIdx.x / Hn;
  const int h = blockIdx.x % Hn;
  const int w = threadIdx.x;

  const size_t cp = ((size_t)b * HW + (size_t)h * Wn + w) * 2;
  const float ix = coords[cp + 0];
  const float iy = coords[cp + 1];

  const float fx0 = floorf(ix), fy0 = floorf(iy);
  const int ix0 = (int)fx0, iy0 = (int)fy0;
  const int ix1 = ix0 + 1,  iy1 = iy0 + 1;
  const float wx1 = ix - fx0, wx0 = 1.0f - wx1;
  const float wy1 = iy - fy0, wy0 = 1.0f - wy1;
  const float w00 = wy0 * wx0, w01 = wy0 * wx1, w10 = wy1 * wx0, w11 = wy1 * wx1;

  const bool vx0 = (ix0 >= 0) & (ix0 < Wn), vx1 = (ix1 >= 0) & (ix1 < Wn);
  const bool vy0 = (iy0 >= 0) & (iy0 < Hn), vy1 = (iy1 >= 0) & (iy1 < Hn);
  const int cx0 = min(max(ix0, 0), Wn - 1), cx1 = min(max(ix1, 0), Wn - 1);
  const int cy0 = min(max(iy0, 0), Hn - 1), cy1 = min(max(iy1, 0), Hn - 1);
  const int o00 = cy0 * Wn + cx0, o01 = cy0 * Wn + cx1;
  const int o10 = cy1 * Wn + cx0, o11 = cy1 * Wn + cx1;
  const float m00 = (vy0 & vx0) ? 1.f : 0.f, m01 = (vy0 & vx1) ? 1.f : 0.f;
  const float m10 = (vy1 & vx0) ? 1.f : 0.f, m11 = (vy1 & vx1) ? 1.f : 0.f;

  const size_t rowoff = (size_t)h * Wn + w;
  const float* xb = x + (size_t)b * Cn * HW;
  float* ob = out + (size_t)b * Cn * HW;
  const float* cab = ca + b * Cn;

#pragma unroll 4
  for (int c = 0; c < Cn; ++c) {
    const float* xc = xb + (size_t)c * HW;
    const float main = xc[rowoff] * cab[c];
    const float v00 = xc[o00] * m00;
    const float v01 = xc[o01] * m01;
    const float v10 = xc[o10] * m10;
    const float v11 = xc[o11] * m11;
    ob[(size_t)c * HW + rowoff] =
        main + (v00 * w00 + v01 * w01 + v10 * w10 + v11 * w11);
  }
}

// ---------------------------------------------------------------------------
extern "C" void kernel_launch(void* const* d_in, const int* in_sizes, int n_in,
                              void* d_out, int out_size, void* d_ws, size_t ws_size,
                              hipStream_t stream) {
  const float* x  = (const float*)d_in[0];
  const float* w1 = (const float*)d_in[1];
  const float* b1 = (const float*)d_in[2];
  const float* w2 = (const float*)d_in[3];
  const float* b2 = (const float*)d_in[4];
  const float* ws = (const float*)d_in[5];
  const float* bs = (const float*)d_in[6];
  float* out = (float*)d_out;

  float* pooled = (float*)d_ws;            // 1024 floats
  float* ca     = pooled + 1024;           // 1024 floats
  float* coords = ca + 1024;               // B*H*W*2 = 1,048,576 floats (4 MB)

  gap_kernel<<<Bn * Cn, 256, 0, stream>>>(x, pooled);
  ca_mlp_kernel<<<1, 32, 0, stream>>>(pooled, w1, b1, w2, b2, ca);
  offset_kernel<<<dim3(Wn / 16, Hn / 16, Bn), dim3(16, 16), 0, stream>>>(x, ws, bs, coords);
  fuse_kernel<<<Bn * Hn, 256, 0, stream>>>(x, ca, coords, out);
}
